// MultiHeadAttention_59622736003670
// MI455X (gfx1250) — matrix operand order repaired and checked by reading, not compiled, not run on hardware
//
#include <hip/hip_runtime.h>

// ---------------------------------------------------------------------------
// MI455X (gfx1250) MHA block: bf16 WMMA GEMMs + flash attention + fused LN.
// wave32, V_WMMA_F32_16X16X32_BF16 for matrix math, async-to-LDS staging
// (ASYNCcnt) for block-shared operand tiles, double-buffered.
// ---------------------------------------------------------------------------

typedef __bf16 bf16_t;
typedef bf16_t v16bf __attribute__((ext_vector_type(16)));
typedef bf16_t v8bf  __attribute__((ext_vector_type(8)));
typedef float  v8f   __attribute__((ext_vector_type(8)));
typedef __attribute__((address_space(3))) bf16_t lds_bf16;

static constexpr int D_MODEL = 1024;
static constexpr int SEQ     = 2048;
static constexpr int BATCH   = 4;
static constexpr int NH      = 16;
static constexpr int DK      = 64;
static constexpr int MTOT    = BATCH * SEQ;   // 8192

#define WMMA_BF16(A, B, C) \
  __builtin_amdgcn_wmma_f32_16x16x32_bf16(false, (A), false, (B), (short)0, (C), false, false)

__device__ __forceinline__ v8f v8f_zero() {
  v8f z = {0.f, 0.f, 0.f, 0.f, 0.f, 0.f, 0.f, 0.f};
  return z;
}

// LDS byte offset of a __shared__-derived pointer (generic -> AS3 -> int).
__device__ __forceinline__ unsigned lds_addr_of(const bf16_t* p) {
  return (unsigned)(uintptr_t)(const lds_bf16*)p;
}

// One 16-byte async global->LDS copy (per lane), tracked with ASYNCcnt.
__device__ __forceinline__ void async_copy_b128(bf16_t* ldsdst, const bf16_t* g) {
  const unsigned loff = lds_addr_of(ldsdst);
  const unsigned long long ga = (unsigned long long)(uintptr_t)g;
  asm volatile("global_load_async_to_lds_b128 %0, %1, off"
               :: "v"(loff), "v"(ga) : "memory");
}

// 64-byte async copy: IOFFSET is added to BOTH the LDS and global address
// (ISA 08_async_tensor §4.4), so one address pair serves all four b128 ops.
__device__ __forceinline__ void async_copy_b128x4(bf16_t* ldsdst, const bf16_t* g) {
  const unsigned loff = lds_addr_of(ldsdst);
  const unsigned long long ga = (unsigned long long)(uintptr_t)g;
  asm volatile(
      "global_load_async_to_lds_b128 %0, %1, off\n\t"
      "global_load_async_to_lds_b128 %0, %1, off offset:16\n\t"
      "global_load_async_to_lds_b128 %0, %1, off offset:32\n\t"
      "global_load_async_to_lds_b128 %0, %1, off offset:48"
      :: "v"(loff), "v"(ga) : "memory");
}

__device__ __forceinline__ void wait_async0() {
  asm volatile("s_wait_asynccnt 0" ::: "memory");
}

// A fragment: 16x32 bf16, row-major source, leading dim `ld`.
// lane m (= lane&15) holds K = kh..kh+7 and kh+16..kh+23, kh = 8*(lane>=16).
__device__ __forceinline__ v16bf load_a_frag(const bf16_t* __restrict__ base, int ld) {
  const int lane = threadIdx.x & 31;
  const int m    = lane & 15;
  const int kh   = (lane >> 4) << 3;
  union { v16bf v; v8bf h[2]; } u;
  u.h[0] = *(const v8bf*)(base + (size_t)m * ld + kh);
  u.h[1] = *(const v8bf*)(base + (size_t)m * ld + kh + 16);
  return u.v;
}

// B fragment: 32x16 bf16, row-major [K][N] source, leading dim `ld`.
// lane n (= lane&15) holds K = kh..kh+7 and kh+16..kh+23, kh = 8*(lane>=16).
__device__ __forceinline__ v16bf load_b_frag(const bf16_t* base, int ld) {
  const int lane = threadIdx.x & 31;
  const int n    = lane & 15;
  const int kh   = (lane >> 4) << 3;
  v16bf v;
#pragma unroll
  for (int i = 0; i < 8; ++i) {
    v[i]     = base[(size_t)(kh + i) * ld + n];
    v[8 + i] = base[(size_t)(kh + 16 + i) * ld + n];
  }
  return v;
}

// ---------------------------------------------------------------------------
__global__ void f32_to_bf16_kernel(const float* __restrict__ src,
                                   bf16_t* __restrict__ dst, int n) {
  int i = blockIdx.x * blockDim.x + threadIdx.x;
  if (i < n) dst[i] = (bf16_t)src[i];
}

// Wt[k][n] = W[n][k]  (bf16) so B-fragments are contiguous loads.
__global__ void transpose_w_kernel(const float* __restrict__ W,
                                   bf16_t* __restrict__ Wt) {
  int i = blockIdx.x * blockDim.x + threadIdx.x;   // i = k*D + n
  int k = i >> 10;
  int n = i & (D_MODEL - 1);
  Wt[i] = (bf16_t)W[(size_t)n * D_MODEL + k];
}

// ---------------------------------------------------------------------------
// Projection GEMM: P = X(bf16, MxK) * Wt(bf16, KxN) + bias.
// Block = 8 waves -> 32x256 tile, wave = 16x64 (4 n-tiles), K-loop step 32.
// Weight tiles (32x256) staged to LDS with double-buffered async copies.
// mode 0: q -> fp32 residual copy + bf16 row-major copy
// mode 1: k -> scatter bf16 into kT[b][h][d][s]
// mode 2: v -> scatter bf16 into vp[b][h][s][d]
__global__ __launch_bounds__(256) void gemm_proj_kernel(
    const bf16_t* __restrict__ X, const bf16_t* __restrict__ Wt,
    const float* __restrict__ bias, int mode,
    float* __restrict__ qf32, bf16_t* __restrict__ qbf,
    bf16_t* __restrict__ scat) {
  __shared__ bf16_t Bs[2][32 * 256];

  const int lane = threadIdx.x & 31;
  const int w    = threadIdx.x >> 5;
  const int arow = blockIdx.x * 32 + (w & 1) * 16;
  const int ncb  = blockIdx.y * 256;         // block n base
  const int nloc = (w >> 1) * 64;            // wave n offset inside block tile

  // stage Wt[ks..ks+32)[ncb..ncb+256) into Bs[buf]: 64B per thread.
  auto stage_b = [&](int ks, int buf) {
    const int row = threadIdx.x >> 3;        // 0..31
    const int col = (threadIdx.x & 7) << 5;  // 0..224 step 32
    async_copy_b128x4(&Bs[buf][row * 256 + col],
                      Wt + (size_t)(ks + row) * D_MODEL + ncb + col);
  };

  v8f acc[4];
#pragma unroll
  for (int t = 0; t < 4; ++t) acc[t] = v8f_zero();

  const bf16_t* a_ptr = X + (size_t)arow * D_MODEL;
  stage_b(0, 0);
  int buf = 0;
  for (int ks = 0; ks < D_MODEL; ks += 32) {
    wait_async0();
    __syncthreads();
    if (ks + 32 < D_MODEL) stage_b(ks + 32, buf ^ 1);
    v16bf a = load_a_frag(a_ptr + ks, D_MODEL);
    __builtin_prefetch(a_ptr + ks + 32, 0, 1);   // global_prefetch_b8
    v16bf b[4];
#pragma unroll
    for (int t = 0; t < 4; ++t) b[t] = load_b_frag(&Bs[buf][nloc + t * 16], 256);
#pragma unroll
    for (int t = 0; t < 4; ++t) acc[t] = WMMA_BF16(a, b[t], acc[t]);
    buf ^= 1;
  }

#pragma unroll
  for (int t = 0; t < 4; ++t) {
    const int col = ncb + nloc + t * 16 + (lane & 15);
    const float bb = bias[col];
#pragma unroll
    for (int r = 0; r < 8; ++r) {
      const int row = arow + r + ((lane >> 4) << 3);
      const float v = acc[t][r] + bb;
      if (mode == 0) {
        qf32[(size_t)row * D_MODEL + col] = v;
        qbf [(size_t)row * D_MODEL + col] = (bf16_t)v;
      } else {
        const int b2 = row >> 11;             // row / SEQ
        const int s = row & (SEQ - 1);
        const int h = col >> 6;               // col / DK
        const int d = col & (DK - 1);
        if (mode == 1)
          scat[((((size_t)b2 * NH + h) * DK + d) * SEQ) + s] = (bf16_t)v;
        else
          scat[((((size_t)b2 * NH + h) * SEQ + s) * DK) + d] = (bf16_t)v;
      }
    }
  }
}

// ---------------------------------------------------------------------------
// Flash attention: block = 8 waves, all sharing one (b,h); each wave owns a
// 16-row Q tile. KV streamed in 32-row tiles staged once per block into LDS
// via double-buffered async copies (8x VMEM reduction vs per-wave loads).
// Per step: 4 score WMMAs, online softmax (xor-shuffle row reductions inside
// the 16-lane C-layout groups), P C-layout -> A-layout via wave-private LDS,
// 4 ctx WMMAs. Scale = 1/sqrt(64) = 0.125 folded in before the max.
__global__ __launch_bounds__(256) void flash_attn_kernel(
    const bf16_t* __restrict__ qbf, const bf16_t* __restrict__ kT,
    const bf16_t* __restrict__ vp, bf16_t* __restrict__ ctx) {
  __shared__ bf16_t Ks[2][64 * 32];   // [d][c], ld=32
  __shared__ bf16_t Vs[2][32 * 64];   // [s][d], ld=64
  __shared__ bf16_t plds[8][16 * 32];

  const int lane = threadIdx.x & 31;
  const int w    = threadIdx.x >> 5;
  const int bh   = blockIdx.x >> 4;                 // 0..63
  const int b    = bh >> 4;
  const int h    = bh & 15;
  const int q0   = ((blockIdx.x & 15) * 8 + w) * 16;

  const bf16_t* qp  = qbf + ((size_t)(b * SEQ + q0)) * D_MODEL + h * DK;
  const bf16_t* kTh = kT + (size_t)bh * DK * SEQ;   // [d][s], ld = SEQ
  const bf16_t* vph = vp + (size_t)bh * SEQ * DK;   // [s][d], ld = DK

  // stage K (64x32) and V (32x64) tiles for kv rows [sc, sc+32)
  auto stage_kv = [&](int sc, int buf) {
    {
      const int row = threadIdx.x >> 2;         // 0..63 (d)
      const int col = (threadIdx.x & 3) << 3;   // 0,8,16,24
      async_copy_b128(&Ks[buf][row * 32 + col], kTh + (size_t)row * SEQ + sc + col);
    }
    {
      const int row = threadIdx.x >> 3;         // 0..31 (s - sc)
      const int col = (threadIdx.x & 7) << 3;   // 0..56 step 8
      async_copy_b128(&Vs[buf][row * 64 + col], vph + (size_t)(sc + row) * DK + col);
    }
  };

  const v16bf aq0 = load_a_frag(qp, D_MODEL);       // d = 0..31
  const v16bf aq1 = load_a_frag(qp + 32, D_MODEL);  // d = 32..63

  float mi[8], li[8];
  v8f acc[4];
#pragma unroll
  for (int r = 0; r < 8; ++r) { mi[r] = -1e30f; li[r] = 0.f; }
#pragma unroll
  for (int t = 0; t < 4; ++t) acc[t] = v8f_zero();

  bf16_t* pl = &plds[w][0];
  const v8f vz = v8f_zero();

  stage_kv(0, 0);
  int buf = 0;
  for (int sj = 0; sj < SEQ; sj += 32) {
    wait_async0();
    __syncthreads();
    if (sj + 32 < SEQ) stage_kv(sj + 32, buf ^ 1);

    // scores tile 16x32 = two 16x16 C tiles, contraction over d = 0..63
    v16bf bk[4];
    bk[0] = load_b_frag(&Ks[buf][0], 32);
    bk[1] = load_b_frag(&Ks[buf][32 * 32], 32);
    bk[2] = load_b_frag(&Ks[buf][16], 32);
    bk[3] = load_b_frag(&Ks[buf][32 * 32 + 16], 32);
    v8f s0 = WMMA_BF16(aq0, bk[0], vz);
    s0 = WMMA_BF16(aq1, bk[1], s0);
    v8f s1 = WMMA_BF16(aq0, bk[2], vz);
    s1 = WMMA_BF16(aq1, bk[3], s1);

#pragma unroll
    for (int r = 0; r < 8; ++r) {
      const float a = s0[r] * 0.125f;
      const float c = s1[r] * 0.125f;
      float t = fmaxf(a, c);
      t = fmaxf(t, __shfl_xor(t, 1, 32));
      t = fmaxf(t, __shfl_xor(t, 2, 32));
      t = fmaxf(t, __shfl_xor(t, 4, 32));
      t = fmaxf(t, __shfl_xor(t, 8, 32));
      const float mnew = fmaxf(mi[r], t);
      const float corr = __expf(mi[r] - mnew);
      const float p0 = __expf(a - mnew);
      const float p1 = __expf(c - mnew);
      float rs = p0 + p1;
      rs += __shfl_xor(rs, 1, 32);
      rs += __shfl_xor(rs, 2, 32);
      rs += __shfl_xor(rs, 4, 32);
      rs += __shfl_xor(rs, 8, 32);
      li[r] = li[r] * corr + rs;
      mi[r] = mnew;
#pragma unroll
      for (int t2 = 0; t2 < 4; ++t2) acc[t2][r] *= corr;
      // C-layout element -> row-major P tile in wave-private LDS (bf16)
      const int row = r + ((lane >> 4) << 3);
      const int col = lane & 15;
      pl[row * 32 + col]      = (bf16_t)p0;
      pl[row * 32 + col + 16] = (bf16_t)p1;
    }
    asm volatile("s_wait_dscnt 0" ::: "memory");

    // reload P in A-fragment layout (K = 32 kv columns)
    union { v16bf v; v8bf hh[2]; } u;
    {
      const int m  = lane & 15;
      const int kh = (lane >> 4) << 3;
      u.hh[0] = *(const v8bf*)(pl + m * 32 + kh);
      u.hh[1] = *(const v8bf*)(pl + m * 32 + kh + 16);
    }

    v16bf bv[4];
#pragma unroll
    for (int t = 0; t < 4; ++t) bv[t] = load_b_frag(&Vs[buf][t * 16], 64);
#pragma unroll
    for (int t = 0; t < 4; ++t) acc[t] = WMMA_BF16(u.v, bv[t], acc[t]);
    buf ^= 1;
  }

#pragma unroll
  for (int r = 0; r < 8; ++r) {
    const float inv = 1.0f / li[r];
    const int row = b * SEQ + q0 + r + ((lane >> 4) << 3);
#pragma unroll
    for (int t = 0; t < 4; ++t) {
      const int col = h * DK + t * 16 + (lane & 15);
      ctx[(size_t)row * D_MODEL + col] = (bf16_t)(acc[t][r] * inv);
    }
  }
}

// ---------------------------------------------------------------------------
// out = LN( ctx @ Wo^T + b_o + q_proj ).  Block = 16 rows x full D.
// Each wave: 16x128 (8 n-tiles, no inter-wave B reuse -> direct global loads,
// W_o stays hot in the 192MB L2).  Row mean/var via shuffle + LDS atomics.
__global__ __launch_bounds__(256) void gemm_out_ln_kernel(
    const bf16_t* __restrict__ ctxb, const bf16_t* __restrict__ Wto,
    const float* __restrict__ bo, const float* __restrict__ qres,
    const float* __restrict__ gamma, const float* __restrict__ beta,
    float* __restrict__ out) {
  __shared__ float rsum[16];
  __shared__ float rsq[16];

  const int lane = threadIdx.x & 31;
  const int w    = threadIdx.x >> 5;
  const int arow = blockIdx.x * 16;
  const int nc0  = w * 128;

  if (threadIdx.x < 16) { rsum[threadIdx.x] = 0.f; rsq[threadIdx.x] = 0.f; }
  __syncthreads();

  v8f acc[8];
#pragma unroll
  for (int t = 0; t < 8; ++t) acc[t] = v8f_zero();

  const bf16_t* a_ptr = ctxb + (size_t)arow * D_MODEL;
  for (int ks = 0; ks < D_MODEL; ks += 32) {
    v16bf a = load_a_frag(a_ptr + ks, D_MODEL);
    __builtin_prefetch(a_ptr + ks + 32, 0, 1);
    v16bf bfr[8];
#pragma unroll
    for (int t = 0; t < 8; ++t)
      bfr[t] = load_b_frag(Wto + (size_t)ks * D_MODEL + nc0 + t * 16, D_MODEL);
#pragma unroll
    for (int t = 0; t < 8; ++t) acc[t] = WMMA_BF16(a, bfr[t], acc[t]);
  }

  float x[8][8];
#pragma unroll
  for (int t = 0; t < 8; ++t) {
    const int col = nc0 + t * 16 + (lane & 15);
    const float bb = bo[col];
#pragma unroll
    for (int r = 0; r < 8; ++r) {
      const int row = arow + r + ((lane >> 4) << 3);
      x[t][r] = acc[t][r] + bb + qres[(size_t)row * D_MODEL + col];
    }
  }

#pragma unroll
  for (int r = 0; r < 8; ++r) {
    float s = 0.f, q = 0.f;
#pragma unroll
    for (int t = 0; t < 8; ++t) { s += x[t][r]; q += x[t][r] * x[t][r]; }
    s += __shfl_xor(s, 1, 32); q += __shfl_xor(q, 1, 32);
    s += __shfl_xor(s, 2, 32); q += __shfl_xor(q, 2, 32);
    s += __shfl_xor(s, 4, 32); q += __shfl_xor(q, 4, 32);
    s += __shfl_xor(s, 8, 32); q += __shfl_xor(q, 8, 32);
    if ((lane & 15) == 0) {
      const int rw = r + ((lane >> 4) << 3);
      atomicAdd(&rsum[rw], s);
      atomicAdd(&rsq[rw], q);
    }
  }
  __syncthreads();

#pragma unroll
  for (int r = 0; r < 8; ++r) {
    const int rw = r + ((lane >> 4) << 3);
    const float mu   = rsum[rw] * (1.0f / D_MODEL);
    const float var  = rsq[rw] * (1.0f / D_MODEL) - mu * mu;
    const float rstd = rsqrtf(var + 1e-5f);
#pragma unroll
    for (int t = 0; t < 8; ++t) {
      const int col = nc0 + t * 16 + (lane & 15);
      out[(size_t)(arow + rw) * D_MODEL + col] =
          (x[t][r] - mu) * rstd * gamma[col] + beta[col];
    }
  }
}

// ---------------------------------------------------------------------------
extern "C" void kernel_launch(void* const* d_in, const int* in_sizes, int n_in,
                              void* d_out, int out_size, void* d_ws, size_t ws_size,
                              hipStream_t stream) {
  (void)in_sizes; (void)n_in; (void)out_size; (void)ws_size;
  const float* Q     = (const float*)d_in[0];
  const float* K     = (const float*)d_in[1];
  const float* V     = (const float*)d_in[2];
  const float* Wq    = (const float*)d_in[3];
  const float* bq    = (const float*)d_in[4];
  const float* Wk    = (const float*)d_in[5];
  const float* bk    = (const float*)d_in[6];
  const float* Wv    = (const float*)d_in[7];
  const float* bv    = (const float*)d_in[8];
  const float* Wo    = (const float*)d_in[9];
  const float* bo    = (const float*)d_in[10];
  const float* gamma = (const float*)d_in[11];
  const float* beta  = (const float*)d_in[12];
  float* out = (float*)d_out;

  const size_t MD = (size_t)MTOT * D_MODEL;     // 8,388,608
  const size_t WD = (size_t)D_MODEL * D_MODEL;  // 1,048,576

  // workspace partition (bf16 region first, fp32 region 16B-aligned after it)
  bf16_t* xq   = (bf16_t*)d_ws;
  bf16_t* xk   = xq + MD;
  bf16_t* xv   = xk + MD;
  bf16_t* Wtq  = xv + MD;
  bf16_t* Wtk  = Wtq + WD;
  bf16_t* Wtv  = Wtk + WD;
  bf16_t* Wto  = Wtv + WD;
  bf16_t* qbf  = Wto + WD;
  bf16_t* kT   = qbf + MD;   // [b][h][d][s]
  bf16_t* vp   = kT + MD;    // [b][h][s][d]
  bf16_t* ctxb = vp + MD;    // [b*s][d]
  float*  qf32 = (float*)(ctxb + MD);

  // 1) fp32 -> bf16 activations
  f32_to_bf16_kernel<<<(int)(MD / 256), 256, 0, stream>>>(Q, xq, (int)MD);
  f32_to_bf16_kernel<<<(int)(MD / 256), 256, 0, stream>>>(K, xk, (int)MD);
  f32_to_bf16_kernel<<<(int)(MD / 256), 256, 0, stream>>>(V, xv, (int)MD);

  // 2) transpose weights to bf16
  transpose_w_kernel<<<(int)(WD / 256), 256, 0, stream>>>(Wq, Wtq);
  transpose_w_kernel<<<(int)(WD / 256), 256, 0, stream>>>(Wk, Wtk);
  transpose_w_kernel<<<(int)(WD / 256), 256, 0, stream>>>(Wv, Wtv);
  transpose_w_kernel<<<(int)(WD / 256), 256, 0, stream>>>(Wo, Wto);

  // 3) projections
  dim3 gproj(MTOT / 32, D_MODEL / 256);
  gemm_proj_kernel<<<gproj, 256, 0, stream>>>(xq, Wtq, bq, 0, qf32, qbf, kT);
  gemm_proj_kernel<<<gproj, 256, 0, stream>>>(xk, Wtk, bk, 1, qf32, qbf, kT);
  gemm_proj_kernel<<<gproj, 256, 0, stream>>>(xv, Wtv, bv, 2, qf32, qbf, vp);

  // 4) flash attention: 64 (b,h) * 16 blocks, 8 waves/block, 16 q-rows/wave
  flash_attn_kernel<<<BATCH * NH * (SEQ / 128), 256, 0, stream>>>(qbf, kT, vp, ctxb);

  // 5) output projection + bias + residual + LayerNorm
  gemm_out_ln_kernel<<<MTOT / 16, 256, 0, stream>>>(ctxb, Wto, bo, qf32, gamma, beta, out);
}
